// MultiHeadAttention_1726576855716
// MI455X (gfx1250) — compile-verified, matrix operand order checked by reading, not run on hardware
//
#include <hip/hip_runtime.h>
#include <hip/hip_bf16.h>
#include <math.h>

// Problem constants (match reference)
#define B_   2
#define C_   512
#define T_   2048
#define H_   8
#define D_   64
#define WIN_ 4
#define BLK_ 256
#define NEG_ (-1e4f)

// Banded attention: per 16-row query tile, band columns <= 16 + 2*BLK = 528.
#define SBW 544   // LDS row stride, multiple of 32 (pads 528 -> 544)

typedef __attribute__((ext_vector_type(16))) _Float16 v16h;
typedef __attribute__((ext_vector_type(8)))  float    v8f;
typedef unsigned int u32x4 __attribute__((ext_vector_type(4)));
typedef int          i32x8 __attribute__((ext_vector_type(8)));
typedef int          i32x4 __attribute__((ext_vector_type(4)));

// ---------------------------------------------------------------------------
// WMMA fragment helpers (CDNA5 ISA 7.12.2, 16-bit 16x32 A layout):
//   lane L: M = L%16, hi = L/16
//   VGPR v (pair e): K = hi*8 + v*2 + e        (v < 4)
//                    K = 16 + hi*8 + (v-4)*2+e (v >= 4)
// B (32x16) mirrored: lane -> N = L%16, same K packing per element.
// C/D (16x16 f32): lane L -> col = L%16, rows (L/16)*8 .. +7 in v8f.
// ---------------------------------------------------------------------------
__device__ __forceinline__ int frag_k(int i, int hi) {
  int v = i >> 1, e = i & 1;
  return (v < 4) ? (hi * 8 + v * 2 + e) : (16 + hi * 8 + (v - 4) * 2 + e);
}

__device__ __forceinline__ v8f wmma_f16(v16h a, v16h b, v8f c) {
  return __builtin_amdgcn_wmma_f32_16x16x32_f16(false, a, false, b, (short)0, c,
                                                false, false);
}

__device__ __forceinline__ v8f zero8() {
  v8f z = {0.f, 0.f, 0.f, 0.f, 0.f, 0.f, 0.f, 0.f};
  return z;
}

// A fragment from row-major [M x K] source, row stride ld.
__device__ __forceinline__ v16h load_a_rm(const _Float16* base, int ld, int m0, int k0) {
  int lane = threadIdx.x & 31;
  int m = m0 + (lane & 15);
  int hi = lane >> 4;
  const _Float16* p = base + (size_t)m * ld + k0;
  v16h a;
#pragma unroll
  for (int i = 0; i < 16; ++i) a[i] = p[frag_k(i, hi)];
  return a;
}

// B fragment from row-major [K x N] source, row stride ld.
__device__ __forceinline__ v16h load_b_rm(const _Float16* base, int ld, int k0, int n0) {
  int lane = threadIdx.x & 31;
  int n = n0 + (lane & 15);
  int hi = lane >> 4;
  v16h b;
#pragma unroll
  for (int i = 0; i < 16; ++i) b[i] = base[(size_t)(k0 + frag_k(i, hi)) * ld + n];
  return b;
}

// B fragment where memory is [N x K] (transposed use: B[k][n] = mem[n*ld + k]).
__device__ __forceinline__ v16h load_b_nk(const _Float16* base, int ld, int k0, int n0) {
  int lane = threadIdx.x & 31;
  int n = n0 + (lane & 15);
  int hi = lane >> 4;
  const _Float16* p = base + (size_t)n * ld + k0;
  v16h b;
#pragma unroll
  for (int i = 0; i < 16; ++i) b[i] = p[frag_k(i, hi)];
  return b;
}

// ---------------------------------------------------------------------------
// CDNA5 async / tensor data movement helpers
// ---------------------------------------------------------------------------
__device__ __forceinline__ unsigned lds_off32(const void* p) {
  // Generic LDS pointers carry the LDS byte offset in the low 32 bits (ISA 10.2).
  return (unsigned)(unsigned long long)p;
}

// One 16B-per-lane async copy: 32 lanes x 16B = 512B per issue (ASYNCcnt +1).
__device__ __forceinline__ void async_b128(unsigned lds_byte, const void* gsrc) {
  unsigned long long ga = (unsigned long long)gsrc;
  asm volatile("global_load_async_to_lds_b128 %0, %1, off"
               :: "v"(lds_byte), "v"(ga)
               : "memory");
}

// Copy `bytes` (multiple of 512) contiguous bytes global->LDS asynchronously.
// Issues bytes/512 async instructions.
__device__ __forceinline__ void async_copy(unsigned lds_byte, const char* g, int bytes) {
  int lane = threadIdx.x & 31;
#pragma unroll
  for (int t = 0; t < bytes / 512; ++t)
    async_b128(lds_byte + (unsigned)(t * 512 + lane * 16), g + t * 512 + lane * 16);
}

__device__ __forceinline__ void wait_async0() { asm volatile("s_wait_asynccnt 0x0" ::: "memory"); }
__device__ __forceinline__ void wait_async4() { asm volatile("s_wait_asynccnt 0x4" ::: "memory"); }
__device__ __forceinline__ void wait_async8() { asm volatile("s_wait_asynccnt 0x8" ::: "memory"); }

// TDM 2-D tile load: f16 elements, tile (tile_d0 x tile_d1), row stride stride0.
// D# per ISA ch.8: group0 {count=1, lds_addr, global_addr, type=2},
// group1 {data_size=2B, tensor dims, tile dims, dim0 stride}.
__device__ __forceinline__ void tdm_load_2d_f16(unsigned lds_byte, const void* gsrc,
                                                unsigned tensor_d0, unsigned tensor_d1,
                                                unsigned tile_d0, unsigned tile_d1,
                                                unsigned stride0) {
  unsigned long long ga = (unsigned long long)gsrc;
  u32x4 g0;
  g0.x = 1u;                                                    // count=1 (user)
  g0.y = lds_byte;                                              // lds_addr
  g0.z = (unsigned)(ga & 0xffffffffu);                          // global_addr lo
  g0.w = (unsigned)((ga >> 32) & 0x01ffffffu) | (2u << 30);     // addr hi | type=2
  i32x8 g1;
  g1[0] = (int)(1u << 16);                                      // data_size=1 (2B)
  g1[1] = (int)((tensor_d0 & 0xffffu) << 16);                   // tensor_dim0 lo16
  g1[2] = (int)((tensor_d0 >> 16) | ((tensor_d1 & 0xffffu) << 16));
  g1[3] = (int)((tensor_d1 >> 16) | (tile_d0 << 16));           // tile_dim0
  g1[4] = (int)(tile_d1 & 0xffffu);                             // tile_dim1, tile_dim2=0
  g1[5] = (int)stride0;                                         // dim0 stride lo32
  g1[6] = 0;
  g1[7] = 0;
  i32x4 gz = {0, 0, 0, 0};
#if __clang_major__ >= 23
  i32x8 gz8 = {0, 0, 0, 0, 0, 0, 0, 0};
  __builtin_amdgcn_tensor_load_to_lds(g0, g1, gz, gz, gz8, 0);
#else
  __builtin_amdgcn_tensor_load_to_lds(g0, g1, gz, gz, 0);
#endif
}

// s_wait_tensorcnt requires a constant-int operand -> literal wrappers.
__device__ __forceinline__ void wait_tensor0() {
  __builtin_amdgcn_s_wait_tensorcnt(0);
  asm volatile("" ::: "memory");
}
__device__ __forceinline__ void wait_tensor1() {
  __builtin_amdgcn_s_wait_tensorcnt(1);
  asm volatile("" ::: "memory");
}

// ---------------------------------------------------------------------------
// Kernel 0: fp32 -> fp16 convert (weights + activations staging)
// ---------------------------------------------------------------------------
__global__ void cvt_f32_to_f16(const float* __restrict__ in, _Float16* __restrict__ out,
                               int n) {
  int i = blockIdx.x * blockDim.x + threadIdx.x;
  if (i < n) out[i] = (_Float16)in[i];
}

// ---------------------------------------------------------------------------
// Kernel 1: Q/K/V projection GEMMs. One wave = one 16x16 output tile.
// B panel (32 x 16 f16) staged by TDM, double-buffered in LDS.
// grid = (T/16, C/16, 3*B)
// ---------------------------------------------------------------------------
__global__ __launch_bounds__(32) void qkv_gemm(
    const _Float16* __restrict__ Wq16, const _Float16* __restrict__ Wk16,
    const _Float16* __restrict__ Wv16, const _Float16* __restrict__ x16,
    const _Float16* __restrict__ c16, const float* __restrict__ bq,
    const float* __restrict__ bk, const float* __restrict__ bv,
    _Float16* __restrict__ qh, _Float16* __restrict__ kh, _Float16* __restrict__ vh) {
  __shared__ __align__(16) _Float16 Bp[2][32 * 16];

  int n0 = blockIdx.x * 16;
  int m0 = blockIdx.y * 16;
  int which = blockIdx.z % 3;
  int b = blockIdx.z / 3;

  const _Float16* W = (which == 0) ? Wq16 : (which == 1) ? Wk16 : Wv16;
  const _Float16* In = ((which == 0) ? x16 : c16) + (size_t)b * C_ * T_;
  const float* bias = (which == 0) ? bq : (which == 1) ? bk : bv;
  _Float16* Out = (which == 0) ? qh : (which == 1) ? kh : vh;
  float scale = (which == 0) ? 0.125f : 1.0f;  // 1/sqrt(64)

  unsigned lds[2] = {lds_off32(&Bp[0][0]), lds_off32(&Bp[1][0])};

  tdm_load_2d_f16(lds[0], In + n0, T_, C_, 16, 32, T_);
  v8f acc = zero8();
#pragma unroll 1
  for (int kk = 0; kk < C_; kk += 32) {
    int buf = (kk >> 5) & 1;
    if (kk + 32 < C_) {
      tdm_load_2d_f16(lds[buf ^ 1], In + (size_t)(kk + 32) * T_ + n0, T_, C_, 16, 32, T_);
      wait_tensor1();
    } else {
      wait_tensor0();
    }
    v16h a = load_a_rm(W, C_, m0, kk);
    v16h bb = load_b_rm(&Bp[buf][0], 16, 0, 0);
    acc = wmma_f16(a, bb, acc);
  }

  int lane = threadIdx.x & 31;
  int col = lane & 15;
  int rbase = (lane >> 4) * 8;
  int t = n0 + col;
#pragma unroll
  for (int r = 0; r < 8; ++r) {
    int o = m0 + rbase + r;
    float v = (acc[r] + bias[o]) * scale;
    int h = o >> 6, d = o & 63;
    Out[(((size_t)b * H_ + h) * T_ + t) * D_ + d] = (_Float16)v;
  }
}

// ---------------------------------------------------------------------------
// Kernel 2: rel-K diagonal dot table: relk[b,h,t,dd] = (q*scale)_t . emb_rel_k[dd]
// ---------------------------------------------------------------------------
__global__ void relk_kernel(const _Float16* __restrict__ qh,
                            const float* __restrict__ erk, float* __restrict__ relk) {
  int idx = blockIdx.x * blockDim.x + threadIdx.x;  // over B*H*T
  if (idx >= B_ * H_ * T_) return;
  const _Float16* qrow = qh + (size_t)idx * D_;
  float qv[D_];
#pragma unroll
  for (int d = 0; d < D_; ++d) qv[d] = (float)qrow[d];
#pragma unroll
  for (int dd = 0; dd < 9; ++dd) {
    float s = 0.f;
#pragma unroll
    for (int d = 0; d < D_; ++d) s += qv[d] * erk[dd * D_ + d];
    relk[(size_t)idx * 9 + dd] = s;
  }
}

// ---------------------------------------------------------------------------
// Kernel 3: fused banded attention. One wave per (b, h, 16-row query tile).
// K / V tiles staged with global_load_async_to_lds_b128 (double-buffered).
// LDS overlay (phases separated by syncs / async waits):
//   [0      .. 34815] : Sl  f32 [16][544]   (phase 1-2)   | phase 3: Vst 8KB + Ol 4KB
//   [34816  .. 52223] : Pl  f16 [16][544]   (phase 2-4)   | phase 1: Kst 4KB
// grid = (T/16, H, B)
// ---------------------------------------------------------------------------
__global__ __launch_bounds__(32) void attn_kernel(
    const _Float16* __restrict__ qh, const _Float16* __restrict__ kh,
    const _Float16* __restrict__ vh, const float* __restrict__ relk,
    const float* __restrict__ erv, const int* __restrict__ attn_mask,
    _Float16* __restrict__ att16) {
  __shared__ __align__(16) char smem[52224];
  float*    Sl  = (float*)smem;                   // [16][SBW]
  _Float16* Pl  = (_Float16*)(smem + 34816);      // [16][SBW]
  _Float16* Kst = (_Float16*)(smem + 34816);      // phase 1: 2 x (16*64) f16
  _Float16* Vst = (_Float16*)smem;                // phase 3: 2 x (32*64) f16
  float*    Ol  = (float*)(smem + 8192);          // phase 3: [16][64]

  int t0 = blockIdx.x * 16;
  int h = blockIdx.y;
  int b = blockIdx.z;
  int lo = t0 - BLK_; if (lo < 0) lo = 0;
  int hi = t0 + 16 + BLK_; if (hi > T_) hi = T_;
  int ncol = hi - lo;
  int lane = threadIdx.x;

  const _Float16* qb = qh + (((size_t)b * H_ + h) * T_) * D_;
  const _Float16* kb = kh + (((size_t)b * H_ + h) * T_) * D_;
  const _Float16* vb = vh + (((size_t)b * H_ + h) * T_) * D_;

  // Query A-fragments (16 rows x D=64 as two K=32 slices), kept in VGPRs.
  v16h qa0 = load_a_rm(qb, D_, t0, 0);
  v16h qa1 = load_a_rm(qb, D_, t0, 32);

  // --- phase 1: scores S[16 x ncol], K tiles async-staged (2KB contiguous) ---
  unsigned kst_lds[2] = {lds_off32(&Kst[0]), lds_off32(&Kst[1024])};
  const char* kbytes = (const char*)(kb + (size_t)lo * D_);
  async_copy(kst_lds[0], kbytes, 2048);
  for (int j = 0; j < ncol; j += 16) {
    int buf = (j >> 4) & 1;
    if (j + 16 < ncol) {
      async_copy(kst_lds[buf ^ 1], kbytes + (size_t)(j + 16) * 128, 2048);
      wait_async4();
    } else {
      wait_async0();
    }
    const _Float16* kt = &Kst[buf * 1024];   // [16 rows s][64 d]
    v8f acc = zero8();
    acc = wmma_f16(qa0, load_b_nk(kt, D_, 0, 0), acc);
    acc = wmma_f16(qa1, load_b_nk(kt, D_, 32, 0), acc);
    int col = lane & 15, rbase = (lane >> 4) * 8;
#pragma unroll
    for (int r = 0; r < 8; ++r) Sl[(rbase + r) * SBW + j + col] = acc[r];
  }
  __syncthreads();

  // --- phase 2: rel-K + mask + softmax: 16 lanes, one band row each ---
  if (lane < 16) {
    int t = t0 + lane;
    const float* rk = relk + ((size_t)((b * H_ + h) * T_) + t) * 9;
#pragma unroll
    for (int dd = 0; dd < 9; ++dd) {
      int s = t + dd - WIN_;
      if (s >= 0 && s < T_) Sl[lane * SBW + (s - lo)] += rk[dd];
    }
    const int* mrow = attn_mask + ((size_t)b * T_ + t) * T_;
    float mx = -3.0e38f;
    for (int cc = 0; cc < ncol; ++cc) {
      int s = lo + cc;
      float v = Sl[lane * SBW + cc];
      int dt = t - s; if (dt < 0) dt = -dt;
      if (dt > BLK_ || mrow[s] == 0) v = NEG_;
      Sl[lane * SBW + cc] = v;
      mx = fmaxf(mx, v);
    }
    float sum = 0.f;
    for (int cc = 0; cc < ncol; ++cc) {
      float e = __expf(Sl[lane * SBW + cc] - mx);
      sum += e;
      Sl[lane * SBW + cc] = e;
    }
    float rinv = 1.0f / sum;
    for (int cc = 0; cc < ncol; ++cc)
      Pl[lane * SBW + cc] = (_Float16)(Sl[lane * SBW + cc] * rinv);
    int cpad = (ncol + 31) & ~31;
    for (int cc = ncol; cc < cpad; ++cc) Pl[lane * SBW + cc] = (_Float16)0.f;
  }
  __syncthreads();

  // --- phase 3: O = P @ V, V chunks (32x64 = 4KB) async-staged ---
  int kchunks = (ncol + 31) >> 5;
  unsigned vst_lds[2] = {lds_off32(&Vst[0]), lds_off32(&Vst[2048])};
  const char* vbytes = (const char*)(vb + (size_t)lo * D_);
  v8f o0 = zero8(), o1 = zero8(), o2 = zero8(), o3 = zero8();
  async_copy(vst_lds[0], vbytes, 4096);
#pragma unroll 1
  for (int kc = 0; kc < kchunks; ++kc) {
    int buf = kc & 1;
    if (kc + 1 < kchunks) {
      async_copy(vst_lds[buf ^ 1], vbytes + (size_t)(kc + 1) * 4096, 4096);
      wait_async8();
    } else {
      wait_async0();
    }
    const _Float16* vt = &Vst[buf * 2048];   // [32 rows s][64 d]
    v16h pa = load_a_rm(Pl, SBW, 0, kc * 32);
    o0 = wmma_f16(pa, load_b_rm(vt, D_, 0, 0), o0);
    o1 = wmma_f16(pa, load_b_rm(vt, D_, 0, 16), o1);
    o2 = wmma_f16(pa, load_b_rm(vt, D_, 0, 32), o2);
    o3 = wmma_f16(pa, load_b_rm(vt, D_, 0, 48), o3);
  }
  {
    int col = lane & 15, rbase = (lane >> 4) * 8;
#pragma unroll
    for (int r = 0; r < 8; ++r) {
      Ol[(rbase + r) * 64 + 0 + col]  = o0[r];
      Ol[(rbase + r) * 64 + 16 + col] = o1[r];
      Ol[(rbase + r) * 64 + 32 + col] = o2[r];
      Ol[(rbase + r) * 64 + 48 + col] = o3[r];
    }
  }
  __syncthreads();

  // --- phase 4: rel-V 9-tap stencil + store merged-head layout [B, C, T] ---
  for (int idx = lane; idx < 16 * 64; idx += 32) {
    int r = idx >> 6, d = idx & 63;
    int t = t0 + r;
    float o = Ol[r * 64 + d];
#pragma unroll
    for (int dd = 0; dd < 9; ++dd) {
      int s = t + dd - WIN_;
      if (s >= 0 && s < T_) o += (float)Pl[r * SBW + (s - lo)] * erv[dd * D_ + d];
    }
    att16[((size_t)b * C_ + h * D_ + d) * T_ + t] = (_Float16)o;
  }
}

// ---------------------------------------------------------------------------
// Kernel 4: output projection: out[b,o,t] = sum_c Wo[o,c]*att[b,c,t] + bo[o]
// B panel staged by TDM, double-buffered. grid = (T/16, C/16, B), f32 output.
// ---------------------------------------------------------------------------
__global__ __launch_bounds__(32) void out_gemm(const _Float16* __restrict__ Wo16,
                                               const _Float16* __restrict__ att16,
                                               const float* __restrict__ bo,
                                               float* __restrict__ out) {
  __shared__ __align__(16) _Float16 Bp[2][32 * 16];

  int n0 = blockIdx.x * 16;
  int m0 = blockIdx.y * 16;
  int b = blockIdx.z;
  const _Float16* In = att16 + (size_t)b * C_ * T_;
  unsigned lds[2] = {lds_off32(&Bp[0][0]), lds_off32(&Bp[1][0])};

  tdm_load_2d_f16(lds[0], In + n0, T_, C_, 16, 32, T_);
  v8f acc = zero8();
#pragma unroll 1
  for (int kk = 0; kk < C_; kk += 32) {
    int buf = (kk >> 5) & 1;
    if (kk + 32 < C_) {
      tdm_load_2d_f16(lds[buf ^ 1], In + (size_t)(kk + 32) * T_ + n0, T_, C_, 16, 32, T_);
      wait_tensor1();
    } else {
      wait_tensor0();
    }
    v16h a = load_a_rm(Wo16, C_, m0, kk);
    v16h bb = load_b_rm(&Bp[buf][0], 16, 0, 0);
    acc = wmma_f16(a, bb, acc);
  }
  int lane = threadIdx.x & 31;
  int col = lane & 15;
  int rbase = (lane >> 4) * 8;
  int t = n0 + col;
#pragma unroll
  for (int r = 0; r < 8; ++r) {
    int o = m0 + rbase + r;
    out[((size_t)b * C_ + o) * T_ + t] = acc[r] + bo[o];
  }
}

// ---------------------------------------------------------------------------
// Launch
// ---------------------------------------------------------------------------
extern "C" void kernel_launch(void* const* d_in, const int* in_sizes, int n_in,
                              void* d_out, int out_size, void* d_ws, size_t ws_size,
                              hipStream_t stream) {
  const float* x = (const float*)d_in[0];
  const float* c = (const float*)d_in[1];
  const int* attn_mask = (const int*)d_in[2];
  const float* Wq = (const float*)d_in[3];
  const float* bq = (const float*)d_in[4];
  const float* Wk = (const float*)d_in[5];
  const float* bk = (const float*)d_in[6];
  const float* Wv = (const float*)d_in[7];
  const float* bv = (const float*)d_in[8];
  const float* Wo = (const float*)d_in[9];
  const float* bo = (const float*)d_in[10];
  const float* erk = (const float*)d_in[11];
  const float* erv = (const float*)d_in[12];
  float* out = (float*)d_out;

  char* ws = (char*)d_ws;
  size_t off = 0;
  auto alloc = [&](size_t bytes) -> void* {
    off = (off + 255) & ~(size_t)255;
    void* p = ws + off;
    off += bytes;
    return p;
  };

  _Float16* Wq16 = (_Float16*)alloc((size_t)C_ * C_ * 2);
  _Float16* Wk16 = (_Float16*)alloc((size_t)C_ * C_ * 2);
  _Float16* Wv16 = (_Float16*)alloc((size_t)C_ * C_ * 2);
  _Float16* Wo16 = (_Float16*)alloc((size_t)C_ * C_ * 2);
  _Float16* x16  = (_Float16*)alloc((size_t)B_ * C_ * T_ * 2);
  _Float16* c16  = (_Float16*)alloc((size_t)B_ * C_ * T_ * 2);
  _Float16* qh   = (_Float16*)alloc((size_t)B_ * H_ * T_ * D_ * 2);
  _Float16* kh   = (_Float16*)alloc((size_t)B_ * H_ * T_ * D_ * 2);
  _Float16* vh   = (_Float16*)alloc((size_t)B_ * H_ * T_ * D_ * 2);
  _Float16* att16 = (_Float16*)alloc((size_t)B_ * C_ * T_ * 2);  // also pads vh overrun
  float* relk = (float*)alloc((size_t)B_ * H_ * T_ * 9 * 4);

  // 0) fp16 staging
  {
    int nW = C_ * C_, nA = B_ * C_ * T_;
    cvt_f32_to_f16<<<(nW + 255) / 256, 256, 0, stream>>>(Wq, Wq16, nW);
    cvt_f32_to_f16<<<(nW + 255) / 256, 256, 0, stream>>>(Wk, Wk16, nW);
    cvt_f32_to_f16<<<(nW + 255) / 256, 256, 0, stream>>>(Wv, Wv16, nW);
    cvt_f32_to_f16<<<(nW + 255) / 256, 256, 0, stream>>>(Wo, Wo16, nW);
    cvt_f32_to_f16<<<(nA + 255) / 256, 256, 0, stream>>>(x, x16, nA);
    cvt_f32_to_f16<<<(nA + 255) / 256, 256, 0, stream>>>(c, c16, nA);
  }

  // 1) Q/K/V projections (WMMA + TDM-staged B panels)
  qkv_gemm<<<dim3(T_ / 16, C_ / 16, 3 * B_), 32, 0, stream>>>(
      Wq16, Wk16, Wv16, x16, c16, bq, bk, bv, qh, kh, vh);

  // 2) rel-K diagonal table
  relk_kernel<<<(B_ * H_ * T_ + 255) / 256, 256, 0, stream>>>(qh, erk, relk);

  // 3) fused banded attention (WMMA + async-to-LDS staged K/V)
  attn_kernel<<<dim3(T_ / 16, H_, B_), 32, 0, stream>>>(qh, kh, vh, relk, erv,
                                                        attn_mask, att16);

  // 4) output projection (WMMA + TDM-staged B panels)
  out_gemm<<<dim3(T_ / 16, C_ / 16, B_), 32, 0, stream>>>(Wo16, att16, bo, out);
}
